// Attention_3126736192252
// MI455X (gfx1250) — matrix operand order repaired and checked by reading, not compiled, not run on hardware
//
#include <hip/hip_runtime.h>

// ---------------------------------------------------------------------------
// Multi-head attention forward, MI455X (gfx1250), bf16 WMMA pipeline + TDM.
//   B=4, N=2048, C=1024, H=16, hd=64
// ---------------------------------------------------------------------------

typedef __attribute__((ext_vector_type(16))) __bf16 bf16x16;
typedef __attribute__((ext_vector_type(8)))  __bf16 bf16x8;
typedef __attribute__((ext_vector_type(8)))  float  f32x8;
typedef unsigned int u32x4 __attribute__((ext_vector_type(4)));
typedef int          i32x4 __attribute__((ext_vector_type(4)));
typedef int          i32x8 __attribute__((ext_vector_type(8)));

constexpr unsigned BB  = 4;
constexpr unsigned SEQ = 2048;   // power of 2 -> shifts
constexpr unsigned CH  = 1024;
constexpr unsigned NH  = 16;
constexpr unsigned HD  = 64;
constexpr unsigned K3  = 3 * CH;          // 3072
constexpr unsigned MROWS = BB * SEQ;      // 8192

#define WMMA_BF16(A, B, C) \
  __builtin_amdgcn_wmma_f32_16x16x32_bf16(false, (A), false, (B), (short)0, (C), false, false)

// pack two f32 -> packed bf16 pair (lets the compiler use v_cvt_pk_bf16_f32)
__device__ inline unsigned pack2bf16(float a, float b) {
  union { __bf16 h[2]; unsigned u; } r;
  r.h[0] = (__bf16)a; r.h[1] = (__bf16)b;
  return r.u;
}

// ----- fragment loaders (contiguous k order: k = 16*(lane/16) + s, same for A and B) ----------------------
__device__ inline bf16x16 load_a_frag(const __bf16* lds, unsigned rowBase,
                                      unsigned ld, unsigned kOff, unsigned lane) {
  unsigned m  = lane & 15u;
  unsigned kh = lane >> 4;
  const __bf16* p = lds + (rowBase + m) * ld + kOff + kh * 16u;
  bf16x8 lo = *(const bf16x8*)p;
  bf16x8 hi = *(const bf16x8*)(p + 8);
  bf16x16 a;
#pragma unroll
  for (int i = 0; i < 8; ++i) { a[i] = lo[i]; a[i + 8] = hi[i]; }
  return a;
}

__device__ inline bf16x16 load_b_frag(const __bf16* ldsBT, unsigned colBase,
                                      unsigned ld, unsigned kOff, unsigned lane) {
  unsigned n  = lane & 15u;
  unsigned kh = lane >> 4;
  const __bf16* p = ldsBT + (colBase + n) * ld + kOff + kh * 16u;
  return *(const bf16x16*)p;
}

// ----- TDM: 2D tile load Global -> LDS (bf16 elements) ----------------------
// tile_x elems per row, tile_y rows, row pitch stride0 (elements).
__device__ inline void tdm_load_bf16_2d(void* ldsDst, const __bf16* gsrc,
                                        unsigned tile_x, unsigned tile_y,
                                        unsigned tensor_x, unsigned tensor_y,
                                        unsigned long long stride0) {
  unsigned long long ga = (unsigned long long)(uintptr_t)gsrc;
  unsigned lds_addr = (unsigned)(uintptr_t)ldsDst;   // LDS offset = low 32 bits
  u32x4 g0;
  g0[0] = 1u;                                        // count=1, user descriptor
  g0[1] = lds_addr;
  g0[2] = (unsigned)(ga & 0xffffffffull);            // global_addr[31:0]
  g0[3] = (unsigned)((ga >> 32) & 0x01ffffffull) | (2u << 30);  // addr[56:32], type=2
  i32x8 g1;
  g1[0] = (int)0x00010000u;                          // wg_mask=0, data_size=1 (2B)
  g1[1] = (int)((tensor_x & 0xffffu) << 16);         // tensor_dim0[15:0]
  g1[2] = (int)((tensor_x >> 16) | ((tensor_y & 0xffffu) << 16));
  g1[3] = (int)((tensor_y >> 16) | (tile_x << 16));  // tile_dim0
  g1[4] = (int)(tile_y);                             // tile_dim1 (tile_dim2=0)
  g1[5] = (int)(unsigned)(stride0 & 0xffffffffull);  // tensor_dim0_stride[31:0]
  g1[6] = (int)(unsigned)((stride0 >> 32) & 0xffffull);
  g1[7] = 0;
  i32x4 z4 = {};
#if defined(__clang_major__) && (__clang_major__ >= 23)
  i32x8 z8 = {};
  __builtin_amdgcn_tensor_load_to_lds(g0, g1, z4, z4, z8, 0);
#else
  __builtin_amdgcn_tensor_load_to_lds(g0, g1, z4, z4, 0);
#endif
}

// ---------------------------------------------------------------------------
// Kernel 1: QKV projection.  out = x @ W_qkv, scattered into
//   q [B,H,N,hd] bf16, k [B,H,N,hd] bf16, vT [B,H,hd,N] bf16
// ---------------------------------------------------------------------------
__global__ void __launch_bounds__(256)
qkv_gemm_kernel(const float* __restrict__ x, const float* __restrict__ W,
                __bf16* __restrict__ q, __bf16* __restrict__ k,
                __bf16* __restrict__ vT) {
  __shared__ __bf16 ldsA[128 * 32];
  __shared__ __bf16 ldsBT[128 * 32];

  const unsigned tid  = threadIdx.x;
  const unsigned lane = tid & 31u;
  const unsigned wid  = tid >> 5;
  const unsigned wm   = wid & 3u;
  const unsigned wn   = wid >> 2;
  const unsigned blockM = blockIdx.y * 128u;
  const unsigned blockN = blockIdx.x * 128u;

  f32x8 acc[2][4] = {};

  for (unsigned k0 = 0; k0 < CH; k0 += 32u) {
    // --- stage A: 128x32 f32 -> packed bf16, two b128 stores/thread ---
    {
      unsigned row = tid >> 1;
      unsigned cb  = (tid & 1u) * 16u;
      const float* src = x + (size_t)(blockM + row) * CH + k0 + cb;
      if (k0 + 32u < CH) __builtin_prefetch(src + 32, 0, 0);
      uint4 p0, p1;
      float4 v0 = *(const float4*)(src + 0);
      float4 v1 = *(const float4*)(src + 4);
      float4 v2 = *(const float4*)(src + 8);
      float4 v3 = *(const float4*)(src + 12);
      p0.x = pack2bf16(v0.x, v0.y); p0.y = pack2bf16(v0.z, v0.w);
      p0.z = pack2bf16(v1.x, v1.y); p0.w = pack2bf16(v1.z, v1.w);
      p1.x = pack2bf16(v2.x, v2.y); p1.y = pack2bf16(v2.z, v2.w);
      p1.z = pack2bf16(v3.x, v3.y); p1.w = pack2bf16(v3.z, v3.w);
      uint4* dst = (uint4*)(ldsA + row * 32u + cb);
      dst[0] = p0; dst[1] = p1;
    }
    // --- stage B^T: pair adjacent K rows so packed store is contiguous ---
    {
      unsigned kr = (tid >> 4) * 2u;        // 0,2,..,30
      unsigned cb = (tid & 15u) * 8u;       // 8 cols per thread
      const float* s0 = W + (size_t)(k0 + kr) * K3 + blockN + cb;
      const float* s1 = s0 + K3;
      if (k0 + 32u < CH) __builtin_prefetch(s0 + 32u * K3, 0, 0);
      float4 a0 = *(const float4*)(s0 + 0), a1 = *(const float4*)(s0 + 4);
      float4 b0 = *(const float4*)(s1 + 0), b1 = *(const float4*)(s1 + 4);
      unsigned* bt = (unsigned*)ldsBT;
      bt[((cb + 0u) * 32u + kr) >> 1] = pack2bf16(a0.x, b0.x);
      bt[((cb + 1u) * 32u + kr) >> 1] = pack2bf16(a0.y, b0.y);
      bt[((cb + 2u) * 32u + kr) >> 1] = pack2bf16(a0.z, b0.z);
      bt[((cb + 3u) * 32u + kr) >> 1] = pack2bf16(a0.w, b0.w);
      bt[((cb + 4u) * 32u + kr) >> 1] = pack2bf16(a1.x, b1.x);
      bt[((cb + 5u) * 32u + kr) >> 1] = pack2bf16(a1.y, b1.y);
      bt[((cb + 6u) * 32u + kr) >> 1] = pack2bf16(a1.z, b1.z);
      bt[((cb + 7u) * 32u + kr) >> 1] = pack2bf16(a1.w, b1.w);
    }
    __syncthreads();

    bf16x16 af0 = load_a_frag(ldsA, wm * 32u + 0u, 32u, 0u, lane);
    bf16x16 af1 = load_a_frag(ldsA, wm * 32u + 16u, 32u, 0u, lane);
#pragma unroll
    for (unsigned sn = 0; sn < 4; ++sn) {
      bf16x16 bf = load_b_frag(ldsBT, wn * 64u + sn * 16u, 32u, 0u, lane);
      acc[0][sn] = WMMA_BF16(af0, bf, acc[0][sn]);
      acc[1][sn] = WMMA_BF16(af1, bf, acc[1][sn]);
    }
    __syncthreads();
  }

  // --- epilogue: scatter into q / k / vT (all unsigned shift/mask math) ---
  const unsigned kh = lane >> 4, ln = lane & 15u;
#pragma unroll
  for (unsigned sm = 0; sm < 2; ++sm)
#pragma unroll
    for (unsigned sn = 0; sn < 4; ++sn)
#pragma unroll
      for (unsigned r = 0; r < 8; ++r) {
        unsigned mg = blockM + wm * 32u + sm * 16u + r + 8u * kh;
        unsigned cg = blockN + wn * 64u + sn * 16u + ln;
        float val = acc[sm][sn][r];
        unsigned b = mg >> 11, n = mg & (SEQ - 1u);
        unsigned three = cg >> 10;
        unsigned cc = cg & (CH - 1u);
        unsigned h = cc >> 6, d = cc & (HD - 1u);
        size_t bh = (size_t)((b << 4) + h);
        __bf16 bv = (__bf16)val;
        if (three == 0u)      q[((bh << 11) + n) * HD + d] = bv;
        else if (three == 1u) k[((bh << 11) + n) * HD + d] = bv;
        else                  vT[((bh << 6) + d) * SEQ + n] = bv;
      }
}

// ---------------------------------------------------------------------------
// Kernel 2: flash attention.  One WG (4 waves) per (b, h, 64-row q block).
// K/V tiles staged via Tensor Data Mover; S = Q K^T and O += P V in WMMA.
// ---------------------------------------------------------------------------
__global__ void __launch_bounds__(128)
flash_attn_kernel(const __bf16* __restrict__ q, const __bf16* __restrict__ k,
                  const __bf16* __restrict__ vT, const int* __restrict__ mask,
                  __bf16* __restrict__ attn) {
  __shared__ __bf16 ldsQ[64 * 64];
  __shared__ __bf16 ldsK[64 * 64];     // K   [kv][d]  (B^T for QK^T)
  __shared__ __bf16 ldsV[64 * 64];     // V^T [d][kv]  (B^T for P V)
  __shared__ float  ldsS[64 * 64];
  __shared__ __bf16 ldsP[64 * 64];
  __shared__ float  addb[64];
  __shared__ float  mrow[64], lrow[64], arow[64];

  const unsigned tid  = threadIdx.x;
  const unsigned lane = tid & 31u;
  const unsigned w    = tid >> 5;
  const unsigned bh   = blockIdx.x;
  const unsigned b    = bh >> 4;       // / NH
  const unsigned h    = bh & 15u;
  const unsigned q0   = blockIdx.y * 64u;
  const unsigned kh   = lane >> 4, ln = lane & 15u;

  const __bf16* qbase = q  + (size_t)bh * SEQ * HD;
  const __bf16* kbase = k  + (size_t)bh * SEQ * HD;
  const __bf16* vbase = vT + (size_t)bh * HD * SEQ;

  // stage Q once (vector copy)
  {
    unsigned row = tid >> 1;
    unsigned db  = (tid & 1u) * 32u;
    const __bf16* src = qbase + (size_t)(q0 + row) * HD + db;
#pragma unroll
    for (int i = 0; i < 32; i += 8)
      *(bf16x8*)(ldsQ + row * 64u + db + i) = *(const bf16x8*)(src + i);
  }
  if (tid < 64u) { mrow[tid] = -1e30f; lrow[tid] = 0.0f; }

  f32x8 Oacc[4] = {};
  const float scale = 0.125f;          // hd^-0.5

  for (unsigned kv0 = 0; kv0 < SEQ; kv0 += 64u) {
    __syncthreads();                   // prior P@V done before K/V overwrite
    if (w == 0u) {
      // TDM: K tile = 64 rows x 64 elems, row pitch HD; V^T tile = 64 x 64, pitch SEQ
      tdm_load_bf16_2d(ldsK, kbase + (size_t)kv0 * HD, HD, 64u, HD, SEQ, HD);
      tdm_load_bf16_2d(ldsV, vbase + kv0, 64u, HD, SEQ, HD, SEQ);
      __builtin_amdgcn_s_wait_tensorcnt(0);
    }
    if (tid < 64u)
      addb[tid] = (mask[(b << 11) + kv0 + tid] != 1) ? -10000.0f : 0.0f;
    __syncthreads();

    // --- S = Q K^T ---
    f32x8 s[4] = {};
#pragma unroll
    for (unsigned ks = 0; ks < 2; ++ks) {
      bf16x16 af = load_a_frag(ldsQ, w * 16u, 64u, ks * 32u, lane);
#pragma unroll
      for (unsigned t = 0; t < 4; ++t) {
        bf16x16 bf = load_b_frag(ldsK, t * 16u, 64u, ks * 32u, lane);
        s[t] = WMMA_BF16(af, bf, s[t]);
      }
    }
#pragma unroll
    for (unsigned t = 0; t < 4; ++t)
#pragma unroll
      for (unsigned r = 0; r < 8; ++r) {
        unsigned rl = w * 16u + r + 8u * kh;
        unsigned cl = t * 16u + ln;
        ldsS[rl * 64u + cl] = s[t][r] * scale + addb[cl];
      }
    __syncthreads();

    // --- online softmax, one row per thread; packed bf16 P stores ---
    if (tid < 64u) {
      float mold = mrow[tid];
      float mx = mold;
#pragma unroll
      for (int j = 0; j < 64; j += 4) {
        float4 v = *(const float4*)(ldsS + tid * 64u + j);
        mx = fmaxf(mx, fmaxf(fmaxf(v.x, v.y), fmaxf(v.z, v.w)));
      }
      float alpha = __expf(mold - mx);
      float sum = 0.0f;
      unsigned* prow = (unsigned*)(ldsP + tid * 64u);
#pragma unroll
      for (int j = 0; j < 64; j += 2) {
        float p0 = __expf(ldsS[tid * 64u + j]     - mx);
        float p1 = __expf(ldsS[tid * 64u + j + 1] - mx);
        sum += p0 + p1;
        prow[j >> 1] = pack2bf16(p0, p1);
      }
      mrow[tid] = mx;
      lrow[tid] = lrow[tid] * alpha + sum;
      arow[tid] = alpha;
    }
    __syncthreads();

    // --- rescale O, accumulate P V ---
#pragma unroll
    for (unsigned r = 0; r < 8; ++r) {
      float alpha = arow[w * 16u + r + 8u * kh];
#pragma unroll
      for (unsigned t = 0; t < 4; ++t) Oacc[t][r] *= alpha;
    }
#pragma unroll
    for (unsigned ks = 0; ks < 2; ++ks) {
      bf16x16 af = load_a_frag(ldsP, w * 16u, 64u, ks * 32u, lane);
#pragma unroll
      for (unsigned t = 0; t < 4; ++t) {
        bf16x16 bf = load_b_frag(ldsV, t * 16u, 64u, ks * 32u, lane);
        Oacc[t] = WMMA_BF16(af, bf, Oacc[t]);
      }
    }
  }
  __syncthreads();

  // --- normalize and write attn [B,N,C] ---
#pragma unroll
  for (unsigned r = 0; r < 8; ++r) {
    unsigned rl = w * 16u + r + 8u * kh;
    float linv = 1.0f / lrow[rl];
    size_t rowOff = ((size_t)((b << 11) + q0 + rl) << 10) + (h << 6);
#pragma unroll
    for (unsigned t = 0; t < 4; ++t)
      attn[rowOff + t * 16u + ln] = (__bf16)(Oacc[t][r] * linv);
  }
}

// ---------------------------------------------------------------------------
// Kernel 3: output projection.  out = attn(bf16) @ W_proj(f32) + b_proj, fp32.
// ---------------------------------------------------------------------------
__global__ void __launch_bounds__(256)
proj_gemm_kernel(const __bf16* __restrict__ A, const float* __restrict__ W,
                 const float* __restrict__ bias, float* __restrict__ out) {
  __shared__ __bf16 ldsA[128 * 32];
  __shared__ __bf16 ldsBT[128 * 32];

  const unsigned tid  = threadIdx.x;
  const unsigned lane = tid & 31u;
  const unsigned wid  = tid >> 5;
  const unsigned wm   = wid & 3u;
  const unsigned wn   = wid >> 2;
  const unsigned blockM = blockIdx.y * 128u;
  const unsigned blockN = blockIdx.x * 128u;

  f32x8 acc[2][4] = {};

  for (unsigned k0 = 0; k0 < CH; k0 += 32u) {
    {
      unsigned row = tid >> 1;
      unsigned cb  = (tid & 1u) * 16u;
      const __bf16* src = A + (size_t)(blockM + row) * CH + k0 + cb;
      if (k0 + 32u < CH) __builtin_prefetch(src + 32, 0, 0);
      *(bf16x8*)(ldsA + row * 32u + cb)      = *(const bf16x8*)(src);
      *(bf16x8*)(ldsA + row * 32u + cb + 8u) = *(const bf16x8*)(src + 8);
    }
    {
      unsigned kr = (tid >> 4) * 2u;
      unsigned cb = (tid & 15u) * 8u;
      const float* s0 = W + (size_t)(k0 + kr) * CH + blockN + cb;
      const float* s1 = s0 + CH;
      if (k0 + 32u < CH) __builtin_prefetch(s0 + 32u * CH, 0, 0);
      float4 a0 = *(const float4*)(s0 + 0), a1 = *(const float4*)(s0 + 4);
      float4 b0 = *(const float4*)(s1 + 0), b1 = *(const float4*)(s1 + 4);
      unsigned* bt = (unsigned*)ldsBT;
      bt[((cb + 0u) * 32u + kr) >> 1] = pack2bf16(a0.x, b0.x);
      bt[((cb + 1u) * 32u + kr) >> 1] = pack2bf16(a0.y, b0.y);
      bt[((cb + 2u) * 32u + kr) >> 1] = pack2bf16(a0.z, b0.z);
      bt[((cb + 3u) * 32u + kr) >> 1] = pack2bf16(a0.w, b0.w);
      bt[((cb + 4u) * 32u + kr) >> 1] = pack2bf16(a1.x, b1.x);
      bt[((cb + 5u) * 32u + kr) >> 1] = pack2bf16(a1.y, b1.y);
      bt[((cb + 6u) * 32u + kr) >> 1] = pack2bf16(a1.z, b1.z);
      bt[((cb + 7u) * 32u + kr) >> 1] = pack2bf16(a1.w, b1.w);
    }
    __syncthreads();

    bf16x16 af0 = load_a_frag(ldsA, wm * 32u + 0u, 32u, 0u, lane);
    bf16x16 af1 = load_a_frag(ldsA, wm * 32u + 16u, 32u, 0u, lane);
#pragma unroll
    for (unsigned sn = 0; sn < 4; ++sn) {
      bf16x16 bf = load_b_frag(ldsBT, wn * 64u + sn * 16u, 32u, 0u, lane);
      acc[0][sn] = WMMA_BF16(af0, bf, acc[0][sn]);
      acc[1][sn] = WMMA_BF16(af1, bf, acc[1][sn]);
    }
    __syncthreads();
  }

  const unsigned kh = lane >> 4, ln = lane & 15u;
#pragma unroll
  for (unsigned sm = 0; sm < 2; ++sm)
#pragma unroll
    for (unsigned sn = 0; sn < 4; ++sn)
#pragma unroll
      for (unsigned r = 0; r < 8; ++r) {
        unsigned mg = blockM + wm * 32u + sm * 16u + r + 8u * kh;
        unsigned cg = blockN + wn * 64u + sn * 16u + ln;
        out[((size_t)mg << 10) + cg] = acc[sm][sn][r] + bias[cg];
      }
}

// ---------------------------------------------------------------------------
extern "C" void kernel_launch(void* const* d_in, const int* in_sizes, int n_in,
                              void* d_out, int out_size, void* d_ws,
                              size_t ws_size, hipStream_t stream) {
  const float* x     = (const float*)d_in[0];
  const int*   mask  = (const int*)d_in[1];
  const float* Wqkv  = (const float*)d_in[2];
  const float* Wproj = (const float*)d_in[3];
  const float* bproj = (const float*)d_in[4];
  float* out = (float*)d_out;

  char* ws = (char*)d_ws;
  const size_t SZ = (size_t)BB * NH * SEQ * HD * sizeof(__bf16);  // 16 MB
  __bf16* q    = (__bf16*)(ws);
  __bf16* k    = (__bf16*)(ws + SZ);
  __bf16* vT   = (__bf16*)(ws + 2 * SZ);
  __bf16* attn = (__bf16*)(ws + 3 * SZ);

  qkv_gemm_kernel<<<dim3(K3 / 128, MROWS / 128), 256, 0, stream>>>(
      x, Wqkv, q, k, vT);
  flash_attn_kernel<<<dim3(BB * NH, SEQ / 64), 128, 0, stream>>>(
      q, k, vT, mask, attn);
  proj_gemm_kernel<<<dim3(CH / 128, MROWS / 128), 256, 0, stream>>>(
      attn, Wproj, bproj, out);
}
